// TransformerCRF_44023414784401
// MI455X (gfx1250) — matrix operand order repaired and checked by reading, not compiled, not run on hardware
//
#include <hip/hip_runtime.h>
#include <hip/hip_bf16.h>
#include <math.h>

// ---------------- problem constants ----------------
#define D_    512
#define FF_   2048
#define H_    8
#define L_    4
#define T_    11
#define B_    64
#define S_    256
#define NTOK  (B_*S_)
#define HD_   64
#define START_ 9
#define STOP_  10
#define NEGI  (-10000.0f)

typedef __bf16 bf16_t;
typedef bf16_t v16bf __attribute__((ext_vector_type(16)));
typedef float  v8f   __attribute__((ext_vector_type(8)));

// CDNA5 async global->LDS staging path (guarded; falls back to sync staging)
#if defined(__gfx1250__) && __has_builtin(__builtin_amdgcn_global_load_async_to_lds_b128) && __has_builtin(__builtin_amdgcn_s_wait_asynccnt)
#define HAVE_ASYNC_LDS 1
#else
#define HAVE_ASYNC_LDS 0
#endif

#if HAVE_ASYNC_LDS
typedef int v4i __attribute__((ext_vector_type(4)));
typedef __attribute__((address_space(1))) v4i glb_v4i;   // global (AS1)
typedef __attribute__((address_space(3))) v4i lds_v4i;   // LDS (AS3)
__device__ __forceinline__ void async_cp16(void* lds, const void* g) {
  __builtin_amdgcn_global_load_async_to_lds_b128((glb_v4i*)g, (lds_v4i*)lds, 0, 0);
}
#endif

__device__ __forceinline__ unsigned short f2bf(float f) {
  unsigned u = __float_as_uint(f);
  unsigned r = ((u >> 16) & 1u) + 0x7FFFu;   // round-to-nearest-even
  return (unsigned short)((u + r) >> 16);
}

// ---------------- fp32 -> bf16 convert ----------------
__global__ void k_cvt_bf16(const float* __restrict__ src,
                           unsigned short* __restrict__ dst, int n) {
  int i = blockIdx.x * blockDim.x + threadIdx.x;
  if (i < n) dst[i] = f2bf(src[i]);
}

// ---------------- embedding + sinusoidal PE ----------------
__global__ void k_embed(const float* __restrict__ emb, const int* __restrict__ sent,
                        float* __restrict__ x32, unsigned short* __restrict__ xb) {
  int tok = blockIdx.x;           // 0..NTOK-1
  int s = tok % S_;
  int id = sent[tok];
  const float scale = 22.627416997969522f;   // sqrt(512)
  for (int d = threadIdx.x; d < D_; d += blockDim.x) {
    int i2 = d >> 1;
    float ang = (float)s * __expf(-(float)(2 * i2) * (9.210340371976184f / (float)D_));
    float pe = (d & 1) ? __cosf(ang) : __sinf(ang);
    float v = emb[(long)id * D_ + d] * scale + pe;
    x32[(long)tok * D_ + d] = v;
    xb[(long)tok * D_ + d] = f2bf(v);
  }
}

// ---------------- generic WMMA bf16 GEMM ----------------
// C[m,n] = sum_k A[m,k] * W[n,k] (+bias[n]); TRANSB: W[n,k] read as Bw[k*ldb+n]
// Block tile 256(M) x 64(N) x 32(K); 8 waves (4x2), each wave 64x32 = 4x2 wmma tiles.
// Double-buffered LDS; async global->LDS staging when available.
template <bool TRANSB, bool RELU>
__global__ __launch_bounds__(256) void k_gemm(
    const unsigned short* __restrict__ A, const unsigned short* __restrict__ Bw,
    const float* __restrict__ bias, float* __restrict__ C, unsigned short* __restrict__ Cb,
    int M, int N, int K, int lda, int ldb, int ldc, int nInner,
    long sAo, long sAi, long sBo, long sBi, long sCo, long sCi)
{
  __shared__ __align__(16) unsigned short As[2][256][40];   // +8 pad, 16B-aligned rows
  __shared__ __align__(16) unsigned short Bs[2][64][40];

  int z  = blockIdx.z;
  int zo = z / nInner, zi = z % nInner;
  const unsigned short* Ab = A  + zo * sAo + zi * sAi;
  const unsigned short* Bb = Bw + zo * sBo + zi * sBi;
  long coff = zo * sCo + zi * sCi;

  int m0 = blockIdx.y * 256;
  int n0 = blockIdx.x * 64;
  int t = threadIdx.x;
  int lane = t & 31, wave = t >> 5;
  int wm = wave & 3, wn = wave >> 2;     // 4x2 wave grid: 64 rows x 32 cols each
  int col = lane & 15, kh = lane >> 4;   // half-K select per ISA 16-bit layouts

  // stage one 256x32 A slab + 64x32 B slab into buffer `buf`
  auto stageA = [&](int k0, int buf) {
#pragma unroll
    for (int i = 0; i < 4; ++i) {
      int chunk = t + i * 256;           // 1024 x 16B chunks
      int r = chunk >> 2, c = chunk & 3;
      const unsigned short* gp = &Ab[(long)(m0 + r) * lda + k0 + c * 8];
#if HAVE_ASYNC_LDS
      async_cp16(&As[buf][r][c * 8], gp);
#else
      *(uint4*)&As[buf][r][c * 8] = *(const uint4*)gp;
#endif
    }
  };
  auto stageB = [&](int k0, int buf) {
    if (!TRANSB) {
      int r = t >> 2, c = t & 3;         // 64 rows x 4 chunks
      const unsigned short* gp = &Bb[(long)(n0 + r) * ldb + k0 + c * 8];
#if HAVE_ASYNC_LDS
      async_cp16(&Bs[buf][r][c * 8], gp);
#else
      *(uint4*)&Bs[buf][r][c * 8] = *(const uint4*)gp;
#endif
    } else {
#pragma unroll
      for (int i = 0; i < 8; ++i) {      // 2048 scalar gathers (strided source)
        int e = t + i * 256;
        int n = e & 63, k = e >> 6;
        Bs[buf][n][k] = Bb[(long)(k0 + k) * ldb + n0 + n];
      }
    }
  };

  v8f acc[4][2] = {};
  const int nIter = K >> 5;

  stageA(0, 0);
  stageB(0, 0);

  for (int it = 0; it < nIter; ++it) {
    int cur = it & 1;
    if (it + 1 < nIter) {
      stageA((it + 1) * 32, cur ^ 1);    // overlap next stage with this compute
      stageB((it + 1) * 32, cur ^ 1);
#if HAVE_ASYNC_LDS
      __builtin_amdgcn_s_wait_asynccnt(TRANSB ? 4 : 5);  // current buffer landed
#endif
    } else {
#if HAVE_ASYNC_LDS
      __builtin_amdgcn_s_wait_asynccnt(0);
#endif
    }
    __syncthreads();

    union Frag { v16bf v; uint4 q[2]; } af[4], bfr[2];
    // A 16x32 bf16: lanes0-15 row M with K 0..7 & 16..23; lanes16-31 K 8..15 & 24..31
#pragma unroll
    for (int i = 0; i < 4; ++i) {
      int r = wm * 64 + i * 16 + col;
      af[i].q[0] = *(const uint4*)&As[cur][r][kh * 8];
      af[i].q[1] = *(const uint4*)&As[cur][r][16 + kh * 8];
    }
    // B 32x16 bf16: lanes0-15 col N with K 0..7 & 16..23; lanes16-31 K 8..15 & 24..31
#pragma unroll
    for (int j = 0; j < 2; ++j) {
      int cn = wn * 32 + j * 16 + col;
      bfr[j].q[0] = *(const uint4*)&Bs[cur][cn][kh * 8];
      bfr[j].q[1] = *(const uint4*)&Bs[cur][cn][16 + kh * 8];
    }
#pragma unroll
    for (int i = 0; i < 4; ++i)
#pragma unroll
      for (int j = 0; j < 2; ++j)
        acc[i][j] = __builtin_amdgcn_wmma_f32_16x16x32_bf16(
            false, af[i].v, false, bfr[j].v, (short)0, acc[i][j], false, false);
    __syncthreads();   // reads done before next overwrite of this buffer
  }

  // epilogue: C/D layout -> row = tile + r + (lane>=16)*8, col = tile + lane%16
#pragma unroll
  for (int i = 0; i < 4; ++i) {
    int rbase = m0 + wm * 64 + i * 16 + kh * 8;
#pragma unroll
    for (int j = 0; j < 2; ++j) {
      int cg = n0 + wn * 32 + j * 16 + col;
      float bv = bias ? bias[cg] : 0.0f;
#pragma unroll
      for (int r = 0; r < 8; ++r) {
        float v = acc[i][j][r] + bv;
        if (RELU) v = fmaxf(v, 0.0f);
        long idx = coff + (long)(rbase + r) * ldc + cg;
        if (C)  C[idx]  = v;
        if (Cb) Cb[idx] = f2bf(v);
      }
    }
  }
}

// ---------------- softmax over keys (wave per row) ----------------
__global__ __launch_bounds__(256) void k_softmax(const float* __restrict__ sc,
                                                 unsigned short* __restrict__ pr,
                                                 const unsigned char* __restrict__ mask) {
  long row = (long)blockIdx.x * 8 + (threadIdx.x >> 5);   // [0, B*H*S)
  int lane = threadIdx.x & 31;
  long bh = row >> 8;
  int b = (int)(bh >> 3);
  const float* p = sc + row * S_;
  float v[8];
  float mx = -1e30f;
#pragma unroll
  for (int i = 0; i < 8; ++i) {
    int c = i * 32 + lane;
    float s = p[c] * 0.125f;                 // 1/sqrt(64)
    v[i] = mask[b * S_ + c] ? s : -1e9f;     // key padding mask
    mx = fmaxf(mx, v[i]);
  }
  for (int o = 16; o > 0; o >>= 1) mx = fmaxf(mx, __shfl_xor(mx, o, 32));
  float sum = 0.f;
#pragma unroll
  for (int i = 0; i < 8; ++i) { v[i] = __expf(v[i] - mx); sum += v[i]; }
  for (int o = 16; o > 0; o >>= 1) sum += __shfl_xor(sum, o, 32);
  float inv = 1.0f / sum;
  unsigned short* q = pr + row * S_;
#pragma unroll
  for (int i = 0; i < 8; ++i) q[i * 32 + lane] = f2bf(v[i] * inv);
}

// ---------------- residual add + LayerNorm (block per row) ----------------
__global__ __launch_bounds__(256) void k_add_ln(const float* __restrict__ x,
                                                const float* __restrict__ d,
                                                const float* __restrict__ g,
                                                const float* __restrict__ bta,
                                                float* __restrict__ xo,
                                                unsigned short* __restrict__ xob) {
  __shared__ float red[16];
  long row = blockIdx.x;
  int t = threadIdx.x;
  float v0 = x[row * D_ + t]       + d[row * D_ + t];
  float v1 = x[row * D_ + t + 256] + d[row * D_ + t + 256];
  float s = v0 + v1, s2 = v0 * v0 + v1 * v1;
  for (int o = 16; o > 0; o >>= 1) { s += __shfl_xor(s, o, 32); s2 += __shfl_xor(s2, o, 32); }
  int lane = t & 31, wave = t >> 5;
  if (lane == 0) { red[wave] = s; red[8 + wave] = s2; }
  __syncthreads();
  if (t == 0) {
    float ts = 0, ts2 = 0;
    for (int i = 0; i < 8; ++i) { ts += red[i]; ts2 += red[8 + i]; }
    red[0] = ts; red[1] = ts2;
  }
  __syncthreads();
  float mean = red[0] * (1.0f / D_);
  float var  = red[1] * (1.0f / D_) - mean * mean;
  float inv  = rsqrtf(var + 1e-5f);
  float y0 = (v0 - mean) * inv * g[t]       + bta[t];
  float y1 = (v1 - mean) * inv * g[t + 256] + bta[t + 256];
  xo[row * D_ + t] = y0;           xo[row * D_ + t + 256] = y1;
  xob[row * D_ + t] = f2bf(y0);    xob[row * D_ + t + 256] = f2bf(y1);
}

// ---------------- emissions: x @ W_tag.T + b_tag (tiny N=11) ----------------
__global__ void k_emissions(const float* __restrict__ x, const float* __restrict__ Wt,
                            const float* __restrict__ bt, float* __restrict__ em) {
  long i = (long)blockIdx.x * blockDim.x + threadIdx.x;
  if (i >= (long)NTOK * T_) return;
  int tok = (int)(i / T_), tg = (int)(i % T_);
  const float* xr = x + (long)tok * D_;
  const float* wr = Wt + (long)tg * D_;
  float acc = bt[tg];
  for (int k = 0; k < D_; ++k) acc += xr[k] * wr[k];
  em[i] = acc;
}

// ---------------- CRF: forward logZ + gold path score ----------------
__global__ void k_zero(float* o) { o[0] = 0.0f; }

__global__ __launch_bounds__(1024) void k_crf(const float* __restrict__ em,
                                              const float* __restrict__ trans,
                                              const int* __restrict__ tags,
                                              const unsigned char* __restrict__ mask,
                                              float* __restrict__ out) {
  __shared__ float tr[T_ * T_];
  int t = threadIdx.x;
  if (t < T_ * T_) tr[t] = trans[t];
  __syncthreads();
  int wave = t >> 5, lane = t & 31;
  int b = blockIdx.x * 32 + wave;          // wave per sequence
  int j = lane < T_ ? lane : 0;
  float alpha = (lane == START_) ? 0.0f : NEGI;
  for (int s = 0; s < S_; ++s) {
    if (!mask[b * S_ + s]) continue;       // wave-uniform
    float mx = -1e30f;
#pragma unroll
    for (int i = 0; i < T_; ++i) {
      float ai = __shfl(alpha, i, 32);
      mx = fmaxf(mx, ai + tr[i * T_ + j]);
    }
    float sum = 0.f;
#pragma unroll
    for (int i = 0; i < T_; ++i) {
      float ai = __shfl(alpha, i, 32);
      sum += __expf(ai + tr[i * T_ + j] - mx);
    }
    float na = mx + __logf(sum) + em[((long)b * S_ + s) * T_ + j];
    if (lane < T_) alpha = na;
  }
  float v = (lane < T_) ? alpha + tr[lane * T_ + STOP_] : -1e30f;
  float mx = v;
  for (int o = 16; o > 0; o >>= 1) mx = fmaxf(mx, __shfl_xor(mx, o, 32));
  float sum = __expf(v - mx);
  for (int o = 16; o > 0; o >>= 1) sum += __shfl_xor(sum, o, 32);
  float logZ = mx + __logf(sum);
  if (lane == 0) {
    float emit = 0.f, trs = 0.f; int prev = START_;
    for (int s = 0; s < S_; ++s) {
      if (!mask[b * S_ + s]) continue;
      int tg = tags[b * S_ + s];
      emit += em[((long)b * S_ + s) * T_ + tg];
      trs  += tr[prev * T_ + tg];
      prev = tg;
    }
    float ts = emit + trs + tr[prev * T_ + STOP_];
    atomicAdd(out, (logZ - ts) * (1.0f / B_));
  }
}

// ---------------- host orchestration ----------------
extern "C" void kernel_launch(void* const* d_in, const int* in_sizes, int n_in,
                              void* d_out, int out_size, void* d_ws, size_t ws_size,
                              hipStream_t stream) {
  (void)in_sizes; (void)n_in; (void)out_size; (void)ws_size;
  const float* emb   = (const float*)d_in[0];
  const float* Wqkv  = (const float*)d_in[1];
  const float* bqkv  = (const float*)d_in[2];
  const float* Wo    = (const float*)d_in[3];
  const float* bo    = (const float*)d_in[4];
  const float* W1    = (const float*)d_in[5];
  const float* b1    = (const float*)d_in[6];
  const float* W2    = (const float*)d_in[7];
  const float* b2    = (const float*)d_in[8];
  const float* ln1g  = (const float*)d_in[9];
  const float* ln1b  = (const float*)d_in[10];
  const float* ln2g  = (const float*)d_in[11];
  const float* ln2b  = (const float*)d_in[12];
  const float* Wtag  = (const float*)d_in[13];
  const float* btag  = (const float*)d_in[14];
  const float* trans = (const float*)d_in[15];
  const int*   sent  = (const int*)d_in[16];
  const int*   tags  = (const int*)d_in[17];
  const unsigned char* amask = (const unsigned char*)d_in[18];
  float* out = (float*)d_out;

  // -------- carve workspace (~380 MB) --------
  char* p = (char*)d_ws;
  auto carve = [&](size_t bytes) -> char* {
    char* r = p; p += (bytes + 255) & ~(size_t)255; return r;
  };
  unsigned short* wqkvb = (unsigned short*)carve((size_t)L_ * 3 * D_ * D_ * 2);
  unsigned short* wob   = (unsigned short*)carve((size_t)L_ * D_ * D_ * 2);
  unsigned short* w1b   = (unsigned short*)carve((size_t)L_ * FF_ * D_ * 2);
  unsigned short* w2b   = (unsigned short*)carve((size_t)L_ * D_ * FF_ * 2);
  float*          x32   = (float*)carve((size_t)NTOK * D_ * 4);
  unsigned short* xb    = (unsigned short*)carve((size_t)NTOK * D_ * 2);
  unsigned short* qkvb  = (unsigned short*)carve((size_t)NTOK * 3 * D_ * 2);
  float*          scor  = (float*)carve((size_t)B_ * H_ * S_ * S_ * 4);
  unsigned short* prob  = (unsigned short*)carve((size_t)B_ * H_ * S_ * S_ * 2);
  unsigned short* ob    = (unsigned short*)carve((size_t)NTOK * D_ * 2);
  float*          tmp32 = (float*)carve((size_t)NTOK * D_ * 4);
  float*          em    = (float*)carve((size_t)NTOK * T_ * 4);
  unsigned short* hb    = (unsigned short*)scor;  // alias: lifetimes disjoint in a layer

  // -------- weight conversion (every call; deterministic) --------
  auto cvt = [&](const float* s, unsigned short* d, int n) {
    k_cvt_bf16<<<(n + 255) / 256, 256, 0, stream>>>(s, d, n);
  };
  cvt(Wqkv, wqkvb, L_ * 3 * D_ * D_);
  cvt(Wo,   wob,   L_ * D_ * D_);
  cvt(W1,   w1b,   L_ * FF_ * D_);
  cvt(W2,   w2b,   L_ * D_ * FF_);

  // -------- embedding + PE --------
  k_embed<<<NTOK, 256, 0, stream>>>(emb, sent, x32, xb);

  const long sBH = (long)S_ * 3 * D_;   // token-row stride within qkv for one batch
  for (int l = 0; l < L_; ++l) {
    // 1) qkv = xb @ Wqkv[l].T + bqkv[l] -> bf16 qkvb  [16384 x 1536]
    k_gemm<false, false><<<dim3(24, 64, 1), 256, 0, stream>>>(
        xb, wqkvb + (long)l * 3 * D_ * D_, bqkv + (long)l * 3 * D_,
        nullptr, qkvb, NTOK, 3 * D_, D_, D_, D_, 3 * D_,
        1, 0, 0, 0, 0, 0, 0);
    // 2) scores = Q @ K.T (batched over B*H)  [256 x 256, K=64]
    k_gemm<false, false><<<dim3(4, 1, B_ * H_), 256, 0, stream>>>(
        qkvb, qkvb + D_, nullptr, scor, nullptr,
        S_, S_, HD_, 3 * D_, 3 * D_, S_,
        H_, sBH, HD_, sBH, HD_, (long)H_ * S_ * S_, (long)S_ * S_);
    // 3) softmax (scale+mask fused) -> bf16 probs
    k_softmax<<<(B_ * H_ * S_) / 8, 256, 0, stream>>>(scor, prob, amask);
    // 4) o = P @ V (batched, transposed-B)  [256 x 64, K=256] -> bf16 ob [B,S,D]
    k_gemm<true, false><<<dim3(1, 1, B_ * H_), 256, 0, stream>>>(
        prob, qkvb + 2 * D_, nullptr, nullptr, ob,
        S_, HD_, S_, S_, 3 * D_, D_,
        H_, (long)H_ * S_ * S_, (long)S_ * S_, sBH, HD_, (long)S_ * D_, HD_);
    // 5) proj = ob @ Wo[l].T + bo[l] -> fp32 tmp32
    k_gemm<false, false><<<dim3(8, 64, 1), 256, 0, stream>>>(
        ob, wob + (long)l * D_ * D_, bo + (long)l * D_,
        tmp32, nullptr, NTOK, D_, D_, D_, D_, D_,
        1, 0, 0, 0, 0, 0, 0);
    // 6) x = LN1(x + proj)
    k_add_ln<<<NTOK, 256, 0, stream>>>(x32, tmp32, ln1g + (long)l * D_, ln1b + (long)l * D_, x32, xb);
    // 7) h = relu(x @ W1[l].T + b1[l]) -> bf16 hb  [16384 x 2048]
    k_gemm<false, true><<<dim3(32, 64, 1), 256, 0, stream>>>(
        xb, w1b + (long)l * FF_ * D_, b1 + (long)l * FF_,
        nullptr, hb, NTOK, FF_, D_, D_, D_, FF_,
        1, 0, 0, 0, 0, 0, 0);
    // 8) f = h @ W2[l].T + b2[l] -> fp32 tmp32
    k_gemm<false, false><<<dim3(8, 64, 1), 256, 0, stream>>>(
        hb, w2b + (long)l * D_ * FF_, b2 + (long)l * D_,
        tmp32, nullptr, NTOK, D_, FF_, FF_, FF_, D_,
        1, 0, 0, 0, 0, 0, 0);
    // 9) x = LN2(x + f)
    k_add_ln<<<NTOK, 256, 0, stream>>>(x32, tmp32, ln2g + (long)l * D_, ln2b + (long)l * D_, x32, xb);
  }

  // -------- emissions + CRF loss --------
  k_emissions<<<(NTOK * T_ + 255) / 256, 256, 0, stream>>>(x32, Wtag, btag, em);
  k_zero<<<1, 1, 0, stream>>>(out);
  k_crf<<<2, 1024, 0, stream>>>(em, trans, tags, amask, out);
}
